// MultiheadAttention_71399536329113
// MI455X (gfx1250) — compile-verified, matrix operand order checked by reading, not run on hardware
//
#include <hip/hip_runtime.h>
#include <stdint.h>

// ---------------------------------------------------------------------------
// Multi-head attention forward for MI455X (gfx1250, wave32, WMMA).
// H=8 heads, D_MODEL=512, D_K=D_V=64, B=8, S=2048.
// Reference softmax is over the QUERY axis (axis=2 of [b,h,q,k]) ->
// per-key-COLUMN normalization -> two-pass (column stats, then ctx).
// All GEMMs: v_wmma_f32_16x16x32_bf16 (fp32 accum).
// Shared operand tiles staged via global_load_async_to_lds_b128 with
// double buffering (ASYNCcnt pipelining).
// ---------------------------------------------------------------------------

#define H_      8
#define DMODEL  512
#define DK      64
#define BATCH   8
#define SEQ     2048
#define NM      (BATCH * SEQ)   // 16384 flattened (b,s) rows

typedef __attribute__((ext_vector_type(16))) __bf16 v16bf;
typedef __attribute__((ext_vector_type(8)))  float  v8f;

struct Frag16 { union { v16bf v; uint32_t u[8]; }; };
struct FragC  { union { v8f   v; float    f[8]; }; };

#define WAIT_ASYNC(n) asm volatile("s_wait_asynccnt " #n ::: "memory")

// one per-lane 16-byte async global->LDS transfer (ASYNCcnt tracked)
__device__ __forceinline__ void async_b128(const uint16_t* gptr, uint16_t* lptr) {
  uint32_t loff = (uint32_t)(uintptr_t)lptr;  // flat->LDS: low 32 bits = LDS byte offset
  asm volatile("global_load_async_to_lds_b128 %0, %1, off"
               :: "v"(loff), "v"((uint64_t)(uintptr_t)gptr)
               : "memory");
}

__device__ __forceinline__ uint16_t f32_to_bf16(float x) {
  uint32_t u = __float_as_uint(x);
  uint32_t r = (u + 0x7FFFu + ((u >> 16) & 1u)) >> 16;  // RNE
  return (uint16_t)r;
}

// 16x32 bf16 fragment loader (wave32 A layout; symmetric B layout when source
// is N-major). base -> (row0, col0) of row-major uint16 matrix, ld elements.
// Compiles to two b128 loads per lane (global or ds depending on source).
__device__ __forceinline__ Frag16 load_frag(const uint16_t* base, int ld) {
  int lane = threadIdx.x & 31;
  int mn = lane & 15, half = lane >> 4;
  const uint16_t* p = base + (size_t)mn * ld;
  Frag16 f;
#pragma unroll
  for (int v = 0; v < 8; ++v) {
    int kb = ((v & 3) << 1) + (half << 3) + ((v >> 2) << 4);  // K base of pair
    f.u[v] = *(const uint32_t*)(p + kb);
  }
  return f;
}

__device__ __forceinline__ void zero_c(FragC& c) {
#pragma unroll
  for (int r = 0; r < 8; ++r) c.f[r] = 0.0f;
}

// ---------------------------------------------------------------------------
// Prep kernels: fp32 -> bf16 conversions, weight transposes.
// ---------------------------------------------------------------------------
__global__ void convert_qkv_bf16(const float* __restrict__ q,
                                 const float* __restrict__ k,
                                 const float* __restrict__ v,
                                 uint16_t* __restrict__ qb,
                                 uint16_t* __restrict__ kb,
                                 uint16_t* __restrict__ vb) {
  size_t n = (size_t)NM * DMODEL;
  size_t stride = (size_t)gridDim.x * blockDim.x;
  for (size_t i = (size_t)blockIdx.x * blockDim.x + threadIdx.x; i < n; i += stride) {
    qb[i] = f32_to_bf16(q[i]);
    kb[i] = f32_to_bf16(k[i]);
    vb[i] = f32_to_bf16(v[i]);
  }
}

// dst[b][c][r] = bf16(src[b][r][c])
__global__ void transpose_w_bf16(const float* __restrict__ src,
                                 uint16_t* __restrict__ dst,
                                 int rows, int cols) {
  const float* s = src + (size_t)blockIdx.y * rows * cols;
  uint16_t*   d = dst + (size_t)blockIdx.y * rows * cols;
  int i = blockIdx.x * blockDim.x + threadIdx.x;
  if (i < rows * cols) {
    int r = i / cols, c = i % cols;
    d[(size_t)c * rows + r] = f32_to_bf16(s[i]);
  }
}

// ---------------------------------------------------------------------------
// Projection GEMM: out[b,h,s,:] = X[b,s,:] @ W[h] + bias[h]
// grid = (NM/64, H, 3) z selects q/k/v. block = 128 (4 waves).
// A (64x32) and B (64x32) tiles staged in LDS via async copies, double
// buffered: chunk c+1 loads overlap chunk c WMMAs (s_wait_asynccnt 4).
// q output pre-scaled by 1/sqrt(DK); v output stored transposed.
// ---------------------------------------------------------------------------
__global__ void proj_kernel(const uint16_t* __restrict__ xq,
                            const uint16_t* __restrict__ xk,
                            const uint16_t* __restrict__ xv,
                            const uint16_t* __restrict__ wqt,
                            const uint16_t* __restrict__ wkt,
                            const uint16_t* __restrict__ wvt,
                            const float* __restrict__ bq,
                            const float* __restrict__ bk,
                            const float* __restrict__ bv,
                            uint16_t* __restrict__ qh,
                            uint16_t* __restrict__ kh,
                            uint16_t* __restrict__ vhT) {
  __shared__ uint16_t sA[2][64 * 32];
  __shared__ uint16_t sB[2][64 * 32];

  int z = blockIdx.z;
  const uint16_t* X  = (z == 0) ? xq  : (z == 1) ? xk  : xv;
  const uint16_t* WT = (z == 0) ? wqt : (z == 1) ? wkt : wvt;
  const float* bias  = (z == 0) ? bq  : (z == 1) ? bk  : bv;

  int h = blockIdx.y;
  int tid = threadIdx.x;
  int w = tid >> 5;
  int lane = tid & 31;
  int gmA = blockIdx.x * 64;
  int gm0 = gmA + w * 16;
  const uint16_t* whead = WT + (size_t)h * DK * DMODEL;

  // stage A(64x32) + B(64x32) bf16 tiles for k-chunk kk into buffer buf.
  // 256 b128 transfers per tile / 128 threads -> 2 per thread per tile.
  auto issue = [&](int kk, int buf) {
#pragma unroll
    for (int j = 0; j < 2; ++j) {
      int id = tid + j * 128;            // 0..255
      int row = id >> 2, seg = id & 3;   // 4 x 16B per 64B row
      async_b128(X + (size_t)(gmA + row) * DMODEL + kk + seg * 8,
                 &sA[buf][id * 8]);
    }
#pragma unroll
    for (int j = 0; j < 2; ++j) {
      int id = tid + j * 128;
      int row = id >> 2, seg = id & 3;
      async_b128(whead + (size_t)row * DMODEL + kk + seg * 8,
                 &sB[buf][id * 8]);
    }
  };

  FragC c[4];
#pragma unroll
  for (int t = 0; t < 4; ++t) zero_c(c[t]);

  issue(0, 0);
  for (int cc = 0; cc < DMODEL / 32; ++cc) {
    if (cc + 1 < DMODEL / 32) {
      issue((cc + 1) * 32, (cc + 1) & 1);
      WAIT_ASYNC(4);   // oldest 4 (chunk cc) complete; next 4 stay in flight
    } else {
      WAIT_ASYNC(0);
    }
    __syncthreads();
    int buf = cc & 1;
    Frag16 a = load_frag(&sA[buf][(w * 16) * 32], 32);
#pragma unroll
    for (int t = 0; t < 4; ++t) {
      Frag16 b = load_frag(&sB[buf][(t * 16) * 32], 32);
      c[t].v = __builtin_amdgcn_wmma_f32_16x16x32_bf16(
          false, a.v, false, b.v, (short)0, c[t].v, false, false);
    }
    __syncthreads();   // reads of buf done before chunk cc+2 overwrites it
  }

  int mloc = lane >> 4;
#pragma unroll
  for (int t = 0; t < 4; ++t) {
    int n = (lane & 15) + t * 16;
    float bi = bias[h * DK + n];
#pragma unroll
    for (int r = 0; r < 8; ++r) {
      float val = c[t].f[r] + bi;
      if (z == 0) val *= 0.125f;              // fold softmax scale into qh
      int gm = gm0 + r + 8 * mloc;
      int b = gm >> 11, s = gm & (SEQ - 1);
      uint16_t bits = f32_to_bf16(val);
      if (z < 2) {
        uint16_t* dst = (z == 0) ? qh : kh;   // [B,H,S,DK]
        dst[((((size_t)b * H_ + h) * SEQ) + s) * DK + n] = bits;
      } else {                                // vhT [B,H,DK,S]
        vhT[((((size_t)b * H_ + h) * DK) + n) * SEQ + s] = bits;
      }
    }
  }
}

// ---------------------------------------------------------------------------
// Pass A: per-key-column softmax stats over the query axis.
// grid = (SEQ/64, H, B), block = 128. Wave w owns 16 key columns; sweeps all
// 2048 queries in 16-row WMMA tiles with online (max, sumexp).
// Direct loads: per-iter working set is tiny and L0-resident.
// ---------------------------------------------------------------------------
__global__ void colstats_kernel(const uint16_t* __restrict__ qh,
                                const uint16_t* __restrict__ kh,
                                float* __restrict__ colmax,
                                float* __restrict__ colsum) {
  int b = blockIdx.z, h = blockIdx.y;
  int w = threadIdx.x >> 5, lane = threadIdx.x & 31;
  size_t bh = (size_t)b * H_ + h;
  const uint16_t* qbh = qh + bh * SEQ * DK;
  const uint16_t* kbh = kh + bh * SEQ * DK;
  int kb0 = blockIdx.x * 64 + w * 16;

  Frag16 bk0 = load_frag(kbh + (size_t)kb0 * DK + 0,  DK);
  Frag16 bk1 = load_frag(kbh + (size_t)kb0 * DK + 32, DK);

  float mrun = -INFINITY, lrun = 0.0f;
  for (int q0 = 0; q0 < SEQ; q0 += 16) {
    Frag16 a0 = load_frag(qbh + (size_t)q0 * DK + 0,  DK);
    Frag16 a1 = load_frag(qbh + (size_t)q0 * DK + 32, DK);
    FragC sc; zero_c(sc);
    sc.v = __builtin_amdgcn_wmma_f32_16x16x32_bf16(false, a0.v, false, bk0.v,
                                                   (short)0, sc.v, false, false);
    sc.v = __builtin_amdgcn_wmma_f32_16x16x32_bf16(false, a1.v, false, bk1.v,
                                                   (short)0, sc.v, false, false);
    float tmax = sc.f[0];
#pragma unroll
    for (int r = 1; r < 8; ++r) tmax = fmaxf(tmax, sc.f[r]);
    tmax = fmaxf(tmax, __shfl_xor(tmax, 16, 32));
    float mnew = fmaxf(mrun, tmax);
    float ps = 0.0f;
#pragma unroll
    for (int r = 0; r < 8; ++r) ps += __expf(sc.f[r] - mnew);
    ps += __shfl_xor(ps, 16, 32);
    lrun = lrun * __expf(mrun - mnew) + ps;
    mrun = mnew;
  }
  if (lane < 16) {
    size_t idx = bh * SEQ + kb0 + lane;
    colmax[idx] = mrun;
    colsum[idx] = lrun;
  }
}

// ---------------------------------------------------------------------------
// Pass B: ctx[q,v] = sum_k exp(s[q,k]-m_k)/l_k * vh[k,v].
// grid = (SEQ/64, H, B), block = 128. Wave w owns 16 q-rows x 64 v-cols.
// kh (32x64) and vhT (64x32) tiles are IDENTICAL across waves -> staged in
// LDS via async copies, double buffered. Score C-tile -> P(bf16) relayout
// through per-wave LDS staging. ctx stored in concat layout [b,s,h*64+v].
// ---------------------------------------------------------------------------
__global__ void ctx_kernel(const uint16_t* __restrict__ qh,
                           const uint16_t* __restrict__ kh,
                           const uint16_t* __restrict__ vhT,
                           const float* __restrict__ colmax,
                           const float* __restrict__ colsum,
                           uint16_t* __restrict__ ctx) {
  __shared__ uint16_t sK[2][32 * 64];   // keys x d
  __shared__ uint16_t sV[2][64 * 32];   // v-rows x keys
  __shared__ uint16_t pst[4][16 * 32];  // per-wave P staging

  int b = blockIdx.z, h = blockIdx.y;
  int tid = threadIdx.x;
  int w = tid >> 5, lane = tid & 31;
  size_t bh = (size_t)b * H_ + h;
  const uint16_t* qbh = qh + bh * SEQ * DK;
  const uint16_t* kbh = kh + bh * SEQ * DK;
  const uint16_t* vbh = vhT + bh * DK * SEQ;
  const float* cm = colmax + bh * SEQ;
  const float* cs = colsum + bh * SEQ;
  int q0 = blockIdx.x * 64 + w * 16;
  int mloc = lane >> 4;

  auto issue = [&](int kc, int buf) {
#pragma unroll
    for (int j = 0; j < 2; ++j) {        // kh tile: 32 rows x 128B
      int id = tid + j * 128;
      int row = id >> 3, seg = id & 7;
      async_b128(kbh + (size_t)(kc + row) * DK + seg * 8, &sK[buf][id * 8]);
    }
#pragma unroll
    for (int j = 0; j < 2; ++j) {        // vhT tile: 64 rows x 64B
      int id = tid + j * 128;
      int row = id >> 2, seg = id & 3;
      async_b128(vbh + (size_t)row * SEQ + kc + seg * 8, &sV[buf][id * 8]);
    }
  };

  Frag16 aq0 = load_frag(qbh + (size_t)q0 * DK + 0,  DK);
  Frag16 aq1 = load_frag(qbh + (size_t)q0 * DK + 32, DK);

  FragC acc[4];
#pragma unroll
  for (int t = 0; t < 4; ++t) zero_c(acc[t]);

  issue(0, 0);
  for (int cc = 0; cc < SEQ / 32; ++cc) {
    int kc = cc * 32;
    if (cc + 1 < SEQ / 32) {
      issue(kc + 32, (cc + 1) & 1);
      WAIT_ASYNC(4);
    } else {
      WAIT_ASYNC(0);
    }
    __syncthreads();                    // staged tiles for chunk cc visible
    int buf = cc & 1;

#pragma unroll
    for (int sub = 0; sub < 2; ++sub) {
      Frag16 b0 = load_frag(&sK[buf][(sub * 16) * DK + 0],  DK);
      Frag16 b1 = load_frag(&sK[buf][(sub * 16) * DK + 32], DK);
      FragC sc; zero_c(sc);
      sc.v = __builtin_amdgcn_wmma_f32_16x16x32_bf16(false, aq0.v, false, b0.v,
                                                     (short)0, sc.v, false, false);
      sc.v = __builtin_amdgcn_wmma_f32_16x16x32_bf16(false, aq1.v, false, b1.v,
                                                     (short)0, sc.v, false, false);
      int n = lane & 15;
      int k0 = kc + sub * 16;
      float m   = cm[k0 + n];
      float inv = 1.0f / cs[k0 + n];
#pragma unroll
      for (int r = 0; r < 8; ++r) {
        float p = __expf(sc.f[r] - m) * inv;
        pst[w][(r + 8 * mloc) * 32 + sub * 16 + n] = f32_to_bf16(p);
      }
    }
    __syncthreads();                    // pstage writes visible cross-lane
    Frag16 ap = load_frag(&pst[w][0], 32);  // P as A-frag (M=q, K=32 keys)
#pragma unroll
    for (int t = 0; t < 4; ++t) {
      Frag16 bv = load_frag(&sV[buf][(t * 16) * 32], 32);
      acc[t].v = __builtin_amdgcn_wmma_f32_16x16x32_bf16(
          false, ap.v, false, bv.v, (short)0, acc[t].v, false, false);
    }
    __syncthreads();                    // buf reads done before cc+2 overwrite
  }

#pragma unroll
  for (int t = 0; t < 4; ++t) {
    int n = (lane & 15) + t * 16;
#pragma unroll
    for (int r = 0; r < 8; ++r) {
      int qg = q0 + r + 8 * mloc;
      ctx[((size_t)b * SEQ + qg) * DMODEL + h * DK + n] = f32_to_bf16(acc[t].f[r]);
    }
  }
}

// ---------------------------------------------------------------------------
// Output projection: out = ctx @ Wo + bo (fp32 out).
// grid = (NM/64, DMODEL/64), block = 128. Same async double-buffered staging.
// ---------------------------------------------------------------------------
__global__ void outproj_kernel(const uint16_t* __restrict__ ctx,
                               const uint16_t* __restrict__ woT,
                               const float* __restrict__ bo,
                               float* __restrict__ out) {
  __shared__ uint16_t sA[2][64 * 32];
  __shared__ uint16_t sB[2][64 * 32];

  int tid = threadIdx.x;
  int w = tid >> 5, lane = tid & 31;
  int gmA = blockIdx.x * 64;
  int gm0 = gmA + w * 16;
  int n0 = blockIdx.y * 64;

  auto issue = [&](int kk, int buf) {
#pragma unroll
    for (int j = 0; j < 2; ++j) {
      int id = tid + j * 128;
      int row = id >> 2, seg = id & 3;
      async_b128(ctx + (size_t)(gmA + row) * DMODEL + kk + seg * 8,
                 &sA[buf][id * 8]);
    }
#pragma unroll
    for (int j = 0; j < 2; ++j) {
      int id = tid + j * 128;
      int row = id >> 2, seg = id & 3;
      async_b128(woT + (size_t)(n0 + row) * DMODEL + kk + seg * 8,
                 &sB[buf][id * 8]);
    }
  };

  FragC c[4];
#pragma unroll
  for (int t = 0; t < 4; ++t) zero_c(c[t]);

  issue(0, 0);
  for (int cc = 0; cc < DMODEL / 32; ++cc) {
    if (cc + 1 < DMODEL / 32) {
      issue((cc + 1) * 32, (cc + 1) & 1);
      WAIT_ASYNC(4);
    } else {
      WAIT_ASYNC(0);
    }
    __syncthreads();
    int buf = cc & 1;
    Frag16 a = load_frag(&sA[buf][(w * 16) * 32], 32);
#pragma unroll
    for (int t = 0; t < 4; ++t) {
      Frag16 b = load_frag(&sB[buf][(t * 16) * 32], 32);
      c[t].v = __builtin_amdgcn_wmma_f32_16x16x32_bf16(
          false, a.v, false, b.v, (short)0, c[t].v, false, false);
    }
    __syncthreads();
  }

  int mloc = lane >> 4;
#pragma unroll
  for (int t = 0; t < 4; ++t) {
    int n = n0 + t * 16 + (lane & 15);
    float bi = bo[n];
#pragma unroll
    for (int r = 0; r < 8; ++r) {
      int gm = gm0 + r + 8 * mloc;
      out[(size_t)gm * DMODEL + n] = c[t].f[r] + bi;
    }
  }
}

// ---------------------------------------------------------------------------
extern "C" void kernel_launch(void* const* d_in, const int* in_sizes, int n_in,
                              void* d_out, int out_size, void* d_ws, size_t ws_size,
                              hipStream_t stream) {
  const float* q  = (const float*)d_in[0];
  const float* k  = (const float*)d_in[1];
  const float* v  = (const float*)d_in[2];
  const float* Wq = (const float*)d_in[3];
  const float* bq = (const float*)d_in[4];
  const float* Wk = (const float*)d_in[5];
  const float* bk = (const float*)d_in[6];
  const float* Wv = (const float*)d_in[7];
  const float* bv = (const float*)d_in[8];
  const float* Wo = (const float*)d_in[9];
  const float* bo = (const float*)d_in[10];
  float* out = (float*)d_out;

  // workspace carve-up (~99 MiB; ctx aliases dead xq buffer)
  char* ws = (char*)d_ws;
  size_t off = 0;
  auto take = [&](size_t bytes) -> char* {
    char* p = ws + off;
    off = (off + bytes + 255) & ~(size_t)255;
    return p;
  };
  uint16_t* xq  = (uint16_t*)take((size_t)NM * DMODEL * 2);
  uint16_t* xk  = (uint16_t*)take((size_t)NM * DMODEL * 2);
  uint16_t* xv  = (uint16_t*)take((size_t)NM * DMODEL * 2);
  uint16_t* wqt = (uint16_t*)take((size_t)H_ * DK * DMODEL * 2);
  uint16_t* wkt = (uint16_t*)take((size_t)H_ * DK * DMODEL * 2);
  uint16_t* wvt = (uint16_t*)take((size_t)H_ * DK * DMODEL * 2);
  uint16_t* wot = (uint16_t*)take((size_t)DMODEL * DMODEL * 2);
  uint16_t* qhb = (uint16_t*)take((size_t)BATCH * H_ * SEQ * DK * 2);
  uint16_t* khb = (uint16_t*)take((size_t)BATCH * H_ * SEQ * DK * 2);
  uint16_t* vht = (uint16_t*)take((size_t)BATCH * H_ * DK * SEQ * 2);
  float* colmax = (float*)take((size_t)BATCH * H_ * SEQ * 4);
  float* colsum = (float*)take((size_t)BATCH * H_ * SEQ * 4);
  uint16_t* ctx = xq;  // reuse: xq is dead after proj_kernel

  convert_qkv_bf16<<<dim3(2048), dim3(256), 0, stream>>>(q, k, v, xq, xk, xv);
  transpose_w_bf16<<<dim3((DMODEL * DK + 255) / 256, H_), dim3(256), 0, stream>>>(Wq, wqt, DMODEL, DK);
  transpose_w_bf16<<<dim3((DMODEL * DK + 255) / 256, H_), dim3(256), 0, stream>>>(Wk, wkt, DMODEL, DK);
  transpose_w_bf16<<<dim3((DMODEL * DK + 255) / 256, H_), dim3(256), 0, stream>>>(Wv, wvt, DMODEL, DK);
  transpose_w_bf16<<<dim3((DMODEL * DMODEL + 255) / 256, 1), dim3(256), 0, stream>>>(Wo, wot, DMODEL, DMODEL);

  proj_kernel<<<dim3(NM / 64, H_, 3), dim3(128), 0, stream>>>(
      xq, xk, xv, wqt, wkt, wvt, bq, bk, bv, qhb, khb, vht);

  colstats_kernel<<<dim3(SEQ / 64, H_, BATCH), dim3(128), 0, stream>>>(
      qhb, khb, colmax, colsum);

  ctx_kernel<<<dim3(SEQ / 64, H_, BATCH), dim3(128), 0, stream>>>(
      qhb, khb, vht, colmax, colsum, ctx);

  outproj_kernel<<<dim3(NM / 64, DMODEL / 64), dim3(128), 0, stream>>>(
      ctx, wot, bo, out);
}
